// DeepseekV2MoE_37271726195197
// MI455X (gfx1250) — matrix operand order repaired and checked by reading, not compiled, not run on hardware
//
#include <hip/hip_runtime.h>
#include <hip/hip_bf16.h>
#include <stdint.h>

// ---------------- problem constants (match reference) ----------------
constexpr int T_  = 1024;   // tokens
constexpr int H_  = 1024;   // hidden
constexpr int I_  = 512;    // moe intermediate
constexpr int E_  = 32;     // experts
constexpr int K_  = 4;      // top-k
constexpr int G_  = 8;      // groups
constexpr int TG_ = 4;      // top-k groups
constexpr int C_  = 256;    // capacity = 2*T*K/E
constexpr int IS_ = 1024;   // shared intermediate = I*NS
constexpr float SCALE_ = 2.5f;

constexpr int LSTR = 40;    // LDS row stride in bf16 elems (32 + 8 pad, 80B)

typedef __attribute__((ext_vector_type(4)))  float  v4f;
typedef __attribute__((ext_vector_type(8)))  float  v8f;
typedef __attribute__((ext_vector_type(16))) __bf16 v16bf;

// pack two f32 -> packed bf16x2 (round-half-up), 3 VALU via v_perm_b32
__device__ inline unsigned bfpack(float x, float y) {
  unsigned xu = __builtin_bit_cast(unsigned, x) + 0x8000u;
  unsigned yu = __builtin_bit_cast(unsigned, y) + 0x8000u;
  return __builtin_amdgcn_perm(yu, xu, 0x07060302u);
}

__device__ inline v8f wmma_bf16(v16bf a, v16bf b, v8f c) {
  return __builtin_amdgcn_wmma_f32_16x16x32_bf16(
      false, a, false, b, (short)0, c, false, false);
}

__device__ inline v16bf make_a(int4 q0, int4 q1) {
  union { int4 q[2]; v16bf v; } a;
  a.q[0] = q0; a.q[1] = q1;
  return a.v;
}
// B tile from LDS (16-bit B 32x16 layout): lane = col (lane&15),
// K values 8*(lane>>4)+0..7 and 16+8*(lane>>4)+0..7.
__device__ inline v16bf load_b_lds(const short* p) {
  union { int4 q[2]; v16bf v; } b;
  b.q[0] = *(const int4*)p;
  b.q[1] = *(const int4*)(p + 16);
  return b.v;
}

// async global(bf16) -> LDS copy, 16B per lane (+optional second 16B chunk).
// the LDS address (VDST) and the global address (VADDR) both add the instruction offset,
// so one shared offset moves both sides. Tracked by ASYNCcnt.
__device__ inline void async_b128(const void* gsrc, void* lds) {
  unsigned dst = (unsigned)(uintptr_t)lds;               // low 32b = LDS off
  unsigned long long src = (unsigned long long)(uintptr_t)gsrc;
  asm volatile("global_load_async_to_lds_b128 %0, %1, off"
               :: "v"(dst), "v"(src) : "memory");
}
__device__ inline void async_b128x2(const void* gsrc, void* lds) {
  unsigned dst = (unsigned)(uintptr_t)lds;
  unsigned long long src = (unsigned long long)(uintptr_t)gsrc;
  asm volatile("global_load_async_to_lds_b128 %0, %1, off\n\t"
               "global_load_async_to_lds_b128 %0, %1, off offset:16"
               :: "v"(dst), "v"(src) : "memory");
}
__device__ inline void wait_async0() {
  asm volatile("s_wait_asynccnt 0x0" ::: "memory");
}

// ---------------- kernel: generic f32 -> bf16 streamer ----------------
__global__ void cvt_f32_bf16_kernel(const float* __restrict__ src,
                                    __bf16* __restrict__ dst, int n4) {
  int i = blockIdx.x * blockDim.x + threadIdx.x;
  if (i < n4) {
    v4f f = ((const v4f*)src)[i];
    int2 q = {(int)bfpack(f.x, f.y), (int)bfpack(f.z, f.w)};
    ((int2*)dst)[i] = q;
  }
}

// ---------------- kernel: init workspace ----------------
__global__ void init_ws_kernel(int* counts, int* tok) {
  for (int i = threadIdx.x; i < E_; i += blockDim.x) counts[i] = 0;
  for (int i = threadIdx.x; i < E_ * C_; i += blockDim.x) tok[i] = 0;
}

// ---------------- kernel: routing (one wave32 per token, lane==expert) ----
__global__ void route_kernel(const float* __restrict__ x,
                             const float* __restrict__ gate_w,
                             const float* __restrict__ e_bias,
                             int* __restrict__ topk_idx,
                             float* __restrict__ topk_w,
                             int* __restrict__ slot_of,
                             int* __restrict__ counts,
                             int* __restrict__ tok) {
  int lane = threadIdx.x & 31;
  int t = blockIdx.x * (blockDim.x >> 5) + (threadIdx.x >> 5);
  const float* xr = x + (size_t)t * H_;
  const float* gr = gate_w + (size_t)lane * H_;
  float logit = 0.f;
  for (int h = 0; h < H_; h += 4) {
    v4f a = *(const v4f*)(xr + h);
    v4f b = *(const v4f*)(gr + h);
    logit += a.x * b.x + a.y * b.y + a.z * b.z + a.w * b.w;
  }
  float score = 1.f / (1.f + __expf(-logit));       // sigmoid
  float sfc = score + e_bias[lane];
  // group score = sum of top-2 within each group of 4 lanes
  int gbase = lane & ~3;
  float v0 = __shfl(sfc, gbase + 0, 32);
  float v1 = __shfl(sfc, gbase + 1, 32);
  float v2 = __shfl(sfc, gbase + 2, 32);
  float v3 = __shfl(sfc, gbase + 3, 32);
  float hi01 = fmaxf(v0, v1), lo01 = fminf(v0, v1);
  float hi23 = fmaxf(v2, v3), lo23 = fminf(v2, v3);
  float gs = fmaxf(hi01, hi23) + fmaxf(fminf(hi01, hi23), fmaxf(lo01, lo23));
  // keep top TG_ groups (tie -> lower index, matching lax.top_k)
  int myg = lane >> 2;
  int grank = 0;
  for (int g = 0; g < G_; ++g) {
    float o = __shfl(gs, g << 2, 32);
    grank += (o > gs) || (o == gs && g < myg);
  }
  float sm = (grank < TG_) ? sfc : 0.f;
  // top-K_ experts wave-wide by rank
  int r = 0;
  for (int l = 0; l < 32; ++l) {
    float o = __shfl(sm, l, 32);
    r += (o > sm) || (o == sm && l < lane);
  }
  bool sel = r < K_;
  float w = sel ? score : 0.f;
  float wsum = w;
  for (int off = 16; off; off >>= 1) wsum += __shfl_xor(wsum, off, 32);
  float wsel = (w / (wsum + 1e-20f)) * SCALE_;
  if (sel) {
    int idx = t * K_ + r;
    topk_idx[idx] = lane;
    topk_w[idx] = wsel;
    int slot = atomicAdd(&counts[lane], 1);
    if (slot < C_) {
      tok[lane * C_ + slot] = t;
      slot_of[idx] = slot;
    } else {
      slot_of[idx] = -1;   // dropped (2x headroom: statistically never)
    }
  }
}

// ============ GEMM1 body: fused gate_up NT-GEMM + SiLU*mul -> bf16 =========
// Block: 256 thr = 8 waves; M=128 (16 rows/wave), N=64 gate + 64 up cols.
// B double-buffered in LDS. f32 B: register stage + bfpack (NT loads).
// bf16 B: GLOBAL_LOAD_ASYNC_TO_LDS_B128 (ASYNCcnt), overlapped with WMMAs.
template <bool GATHER, bool BF16B>
__device__ __forceinline__ void gemm1_body(
    const __bf16* __restrict__ abf,    // (T,H) bf16 token rows
    const void* __restrict__ wsrc_v,   // (2*UP, H) f32 or bf16 rows
    const int* __restrict__ toklist,   // GATHER ? C_ tokens : nullptr
    __bf16* __restrict__ actdst,       // (Mtot, UP)
    int up_off) {                      // UP (= I_ or IS_), also out row stride
  __shared__ __align__(16) short lb[2][128 * LSTR];
  int tid = threadIdx.x;
  int wave = tid >> 5, lane = tid & 31;
  int m0 = blockIdx.x * 128 + wave * 16;
  int n0 = blockIdx.y * 64;

  // --- staging role: 2 threads per staged row, 16 K-values each ---
  int srow = tid >> 1;                 // 0..127 (0-63 gate, 64-127 up)
  int khalf = (tid & 1) * 16;
  int wrow = (srow < 64) ? (n0 + srow) : (up_off + n0 + (srow - 64));
  const float*  gsrc_f = (const float*)wsrc_v + (size_t)wrow * H_ + khalf;
  const __bf16* gsrc_h = (const __bf16*)wsrc_v + (size_t)wrow * H_ + khalf;
  short* ld0 = &lb[0][srow * LSTR + khalf];
  short* ld1 = &lb[1][srow * LSTR + khalf];

  // --- A role ---
  int row = m0 + (lane & 15);
  int tokrow;
  if constexpr (GATHER) tokrow = toklist[row]; else tokrow = row;
  const __bf16* arow = abf + (size_t)tokrow * H_ + ((lane >> 4) * 8);
  int colb = (lane & 15);
  int bk0e = (lane >> 4) * 8;

  v8f accG[4], accU[4];
#pragma unroll
  for (int tt = 0; tt < 4; ++tt) { accG[tt] = 0.f; accU[tt] = 0.f; }

  v4f f0, f1, f2, f3;                  // f32 B register stage
  int4 qa0, qa1;                       // A prefetch

  // prologue: start B stage for step 0
  if constexpr (BF16B) {
    async_b128x2(gsrc_h, ld0);
  } else {
    const v4f* p = (const v4f*)gsrc_f;
    f0 = __builtin_nontemporal_load(p);
    f1 = __builtin_nontemporal_load(p + 1);
    f2 = __builtin_nontemporal_load(p + 2);
    f3 = __builtin_nontemporal_load(p + 3);
  }
  qa0 = *(const int4*)arow;
  qa1 = *(const int4*)(arow + 16);
  if constexpr (BF16B) {
    wait_async0();
  } else {
    int4 q0 = {(int)bfpack(f0.x, f0.y), (int)bfpack(f0.z, f0.w),
               (int)bfpack(f1.x, f1.y), (int)bfpack(f1.z, f1.w)};
    int4 q1 = {(int)bfpack(f2.x, f2.y), (int)bfpack(f2.z, f2.w),
               (int)bfpack(f3.x, f3.y), (int)bfpack(f3.z, f3.w)};
    ((int4*)ld0)[0] = q0; ((int4*)ld0)[1] = q1;
  }
  __syncthreads();

  constexpr int NSTEP = H_ / 32;
  for (int s = 0; s < NSTEP; ++s) {
    int p = s & 1;
    v16bf av = make_a(qa0, qa1);
    bool pf = (s + 1) < NSTEP;
    if (pf) {
      if constexpr (BF16B) {
        async_b128x2(gsrc_h + (size_t)(s + 1) * 32, p ? ld0 : ld1);
      } else {
        const v4f* bp = (const v4f*)(gsrc_f + (s + 1) * 32);
        f0 = __builtin_nontemporal_load(bp);
        f1 = __builtin_nontemporal_load(bp + 1);
        f2 = __builtin_nontemporal_load(bp + 2);
        f3 = __builtin_nontemporal_load(bp + 3);
      }
      qa0 = *(const int4*)(arow + (s + 1) * 32);
      qa1 = *(const int4*)(arow + (s + 1) * 32 + 16);
    }
    // tile order g0,u0,g1,u1,g2,u2,g3,u3
    const short* lbp = lb[p];
    auto tilep = [&](int i) -> const short* {
      int r = ((i & 1) ? 64 : 0) + (i >> 1) * 16 + colb;
      return lbp + r * LSTR + bk0e;
    };
    v16bf bcur = load_b_lds(tilep(0));
    v16bf bnxt;
#pragma unroll
    for (int i = 0; i < 8; ++i) {
      if (i < 7) bnxt = load_b_lds(tilep(i + 1));
      if (i & 1) accU[i >> 1] = wmma_bf16(av, bcur, accU[i >> 1]);
      else       accG[i >> 1] = wmma_bf16(av, bcur, accG[i >> 1]);
      bcur = bnxt;
    }
    if constexpr (BF16B) {
      wait_async0();                 // transfer ran concurrently with WMMAs
    } else if (pf) {
      short* ld = p ? ld0 : ld1;
      int4 q0 = {(int)bfpack(f0.x, f0.y), (int)bfpack(f0.z, f0.w),
                 (int)bfpack(f1.x, f1.y), (int)bfpack(f1.z, f1.w)};
      int4 q1 = {(int)bfpack(f2.x, f2.y), (int)bfpack(f2.z, f2.w),
                 (int)bfpack(f3.x, f3.y), (int)bfpack(f3.z, f3.w)};
      ((int4*)ld)[0] = q0; ((int4*)ld)[1] = q1;
    }
    __syncthreads();
  }

  // epilogue: act = silu(gate) * up -> bf16
#pragma unroll
  for (int tt = 0; tt < 4; ++tt)
#pragma unroll
    for (int rr = 0; rr < 8; ++rr) {
      float g = accG[tt][rr], u = accU[tt][rr];
      float a = (g / (1.f + __expf(-g))) * u;
      int m = m0 + rr + 8 * (lane >> 4);
      int n = n0 + tt * 16 + (lane & 15);
      actdst[(size_t)m * up_off + n] = (__bf16)a;
    }
}

// ============ GEMM2 body: act(bf16) @ W_down^T -> f32 ======================
template <int KDIM, bool BF16B>
__device__ __forceinline__ void gemm2_body(
    const __bf16* __restrict__ asrc,   // (Mtot, KDIM) bf16
    const void* __restrict__ wsrc_v,   // (H_, KDIM) f32 or bf16
    float* __restrict__ ydst,          // (Mtot, H_)
    bool nt_out) {
  __shared__ __align__(16) short lb[2][64 * LSTR];
  int tid = threadIdx.x;
  int wave = tid >> 5, lane = tid & 31;
  int m0 = blockIdx.x * 128 + wave * 16;
  int n0 = blockIdx.y * 64;

  // staging: 4 threads per row, 8 K-values each
  int srow = tid >> 2;                 // 0..63
  int kq = (tid & 3) * 8;
  const float*  gsrc_f = (const float*)wsrc_v + (size_t)(n0 + srow) * KDIM + kq;
  const __bf16* gsrc_h = (const __bf16*)wsrc_v + (size_t)(n0 + srow) * KDIM + kq;
  short* ld0 = &lb[0][srow * LSTR + kq];
  short* ld1 = &lb[1][srow * LSTR + kq];

  const __bf16* arow = asrc + (size_t)(m0 + (lane & 15)) * KDIM + ((lane >> 4) * 8);
  int colb = (lane & 15);
  int bk0e = (lane >> 4) * 8;

  v8f acc[4];
#pragma unroll
  for (int tt = 0; tt < 4; ++tt) acc[tt] = 0.f;

  v4f f0, f1;
  int4 qa0, qa1;

  if constexpr (BF16B) {
    async_b128(gsrc_h, ld0);
  } else {
    const v4f* p = (const v4f*)gsrc_f;
    f0 = __builtin_nontemporal_load(p);
    f1 = __builtin_nontemporal_load(p + 1);
  }
  qa0 = *(const int4*)arow;
  qa1 = *(const int4*)(arow + 16);
  if constexpr (BF16B) {
    wait_async0();
  } else {
    int4 q = {(int)bfpack(f0.x, f0.y), (int)bfpack(f0.z, f0.w),
              (int)bfpack(f1.x, f1.y), (int)bfpack(f1.z, f1.w)};
    ((int4*)ld0)[0] = q;
  }
  __syncthreads();

  constexpr int NSTEP = KDIM / 32;
  for (int s = 0; s < NSTEP; ++s) {
    int p = s & 1;
    v16bf av = make_a(qa0, qa1);
    bool pf = (s + 1) < NSTEP;
    if (pf) {
      if constexpr (BF16B) {
        async_b128(gsrc_h + (size_t)(s + 1) * 32, p ? ld0 : ld1);
      } else {
        const v4f* bp = (const v4f*)(gsrc_f + (s + 1) * 32);
        f0 = __builtin_nontemporal_load(bp);
        f1 = __builtin_nontemporal_load(bp + 1);
      }
      qa0 = *(const int4*)(arow + (s + 1) * 32);
      qa1 = *(const int4*)(arow + (s + 1) * 32 + 16);
    }
    const short* lbp = lb[p];
    auto tilep = [&](int i) -> const short* {
      return lbp + (i * 16 + colb) * LSTR + bk0e;
    };
    v16bf bcur = load_b_lds(tilep(0));
    v16bf bnxt;
#pragma unroll
    for (int i = 0; i < 4; ++i) {
      if (i < 3) bnxt = load_b_lds(tilep(i + 1));
      acc[i] = wmma_bf16(av, bcur, acc[i]);
      bcur = bnxt;
    }
    if constexpr (BF16B) {
      wait_async0();
    } else if (pf) {
      short* ld = p ? ld0 : ld1;
      int4 q = {(int)bfpack(f0.x, f0.y), (int)bfpack(f0.z, f0.w),
                (int)bfpack(f1.x, f1.y), (int)bfpack(f1.z, f1.w)};
      ((int4*)ld)[0] = q;
    }
    __syncthreads();
  }

#pragma unroll
  for (int tt = 0; tt < 4; ++tt)
#pragma unroll
    for (int rr = 0; rr < 8; ++rr) {
      int m = m0 + rr + 8 * (lane >> 4);
      int n = n0 + tt * 16 + (lane & 15);
      float v = acc[tt][rr];
      float* dp = &ydst[(size_t)m * H_ + n];
      if (nt_out) __builtin_nontemporal_store(v, dp);
      else *dp = v;
    }
}

// ---------------- GEMM kernel wrappers ----------------
__global__ void expert_gemm1_kernel(const __bf16* __restrict__ xbf,
                                    const float* __restrict__ wgu,
                                    const int* __restrict__ tok,
                                    __bf16* __restrict__ act) {
  int e = blockIdx.z;
  gemm1_body<true, false>(xbf, wgu + (size_t)e * 2 * I_ * H_, tok + e * C_,
                          act + (size_t)e * C_ * I_, I_);
}
__global__ void shared_gemm1_kernel(const __bf16* __restrict__ xbf,
                                    const __bf16* __restrict__ wgus_bf,
                                    __bf16* __restrict__ acts) {
  gemm1_body<false, true>(xbf, wgus_bf, nullptr, acts, IS_);
}
__global__ void expert_gemm2_kernel(const __bf16* __restrict__ act,
                                    const float* __restrict__ wd,
                                    float* __restrict__ ye) {
  int e = blockIdx.z;
  gemm2_body<I_, false>(act + (size_t)e * C_ * I_, wd + (size_t)e * H_ * I_,
                        ye + (size_t)e * C_ * H_, /*nt_out=*/true);
}
__global__ void shared_gemm2_kernel(const __bf16* __restrict__ acts,
                                    const __bf16* __restrict__ wds_bf,
                                    float* __restrict__ out) {
  gemm2_body<IS_, true>(acts, wds_bf, out, /*nt_out=*/false);
}

// ---------------- kernel: combine routed contributions into d_out -------
__global__ void combine_kernel(float* __restrict__ out,
                               const float* __restrict__ ye,
                               const int* __restrict__ topk_idx,
                               const float* __restrict__ topk_w,
                               const int* __restrict__ slot_of) {
  int t = blockIdx.x;
  int h4 = threadIdx.x;                 // H/4 = 256 float4 per row
  v4f* op = (v4f*)(out + (size_t)t * H_);
  v4f acc = op[h4];
#pragma unroll
  for (int k = 0; k < K_; ++k) {
    int e = topk_idx[t * K_ + k];
    int s = slot_of[t * K_ + k];
    float w = topk_w[t * K_ + k];
    if (s >= 0) {
      const v4f* yp = (const v4f*)(ye + ((size_t)e * C_ + s) * H_) + h4;
      v4f y = __builtin_nontemporal_load(yp);
      acc.x += w * y.x; acc.y += w * y.y; acc.z += w * y.z; acc.w += w * y.w;
    }
  }
  op[h4] = acc;
}

// ---------------- launch ----------------
extern "C" void kernel_launch(void* const* d_in, const int* in_sizes, int n_in,
                              void* d_out, int out_size, void* d_ws, size_t ws_size,
                              hipStream_t stream) {
  const float* x    = (const float*)d_in[0];
  const float* gw   = (const float*)d_in[1];
  const float* eb   = (const float*)d_in[2];
  const float* wgu  = (const float*)d_in[3];
  const float* wd   = (const float*)d_in[4];
  const float* wgus = (const float*)d_in[5];
  const float* wds  = (const float*)d_in[6];
  float* out = (float*)d_out;

  // workspace layout (~52 MB)
  char* base = (char*)d_ws;
  size_t off = 0;
  auto carve = [&](size_t bytes) -> char* {
    char* p = base + off;
    off = (off + bytes + 255) & ~(size_t)255;
    return p;
  };
  int*    counts   = (int*)   carve(E_ * sizeof(int));
  int*    tok      = (int*)   carve((size_t)E_ * C_ * sizeof(int));
  int*    topk_idx = (int*)   carve((size_t)T_ * K_ * sizeof(int));
  float*  topk_w   = (float*) carve((size_t)T_ * K_ * sizeof(float));
  int*    slot_of  = (int*)   carve((size_t)T_ * K_ * sizeof(int));
  __bf16* xbf      = (__bf16*)carve((size_t)T_ * H_ * sizeof(__bf16));
  __bf16* wgus_bf  = (__bf16*)carve((size_t)2 * IS_ * H_ * sizeof(__bf16));
  __bf16* wds_bf   = (__bf16*)carve((size_t)H_ * IS_ * sizeof(__bf16));
  __bf16* act_e    = (__bf16*)carve((size_t)E_ * C_ * I_ * sizeof(__bf16));
  __bf16* acts_sh  = (__bf16*)carve((size_t)T_ * IS_ * sizeof(__bf16));
  float*  ye       = (float*) carve((size_t)E_ * C_ * H_ * sizeof(float));
  (void)ws_size; (void)n_in; (void)in_sizes; (void)out_size;

  init_ws_kernel<<<1, 256, 0, stream>>>(counts, tok);
  // pre-conversions (bf16 A everywhere; bf16 B for shared-expert GEMMs)
  cvt_f32_bf16_kernel<<<(T_ * H_ / 4 + 255) / 256, 256, 0, stream>>>(
      x, xbf, T_ * H_ / 4);
  cvt_f32_bf16_kernel<<<(2 * IS_ * H_ / 4 + 255) / 256, 256, 0, stream>>>(
      wgus, wgus_bf, 2 * IS_ * H_ / 4);
  cvt_f32_bf16_kernel<<<(H_ * IS_ / 4 + 255) / 256, 256, 0, stream>>>(
      wds, wds_bf, H_ * IS_ / 4);
  route_kernel<<<T_ / 8, 256, 0, stream>>>(x, gw, eb, topk_idx, topk_w,
                                           slot_of, counts, tok);
  expert_gemm1_kernel<<<dim3(C_ / 128, I_ / 64, E_), 256, 0, stream>>>(
      xbf, wgu, tok, act_e);
  expert_gemm2_kernel<<<dim3(C_ / 128, H_ / 64, E_), 256, 0, stream>>>(
      act_e, wd, ye);
  shared_gemm1_kernel<<<dim3(T_ / 128, IS_ / 64), 256, 0, stream>>>(
      xbf, wgus_bf, acts_sh);
  shared_gemm2_kernel<<<dim3(T_ / 128, H_ / 64), 256, 0, stream>>>(
      acts_sh, wds_bf, out);
  combine_kernel<<<T_, 256, 0, stream>>>(out, ye, topk_idx, topk_w, slot_of);
}
